// RNN_classifier_21414706938182
// MI455X (gfx1250) — matrix operand order repaired and checked by reading, not compiled, not run on hardware
//
#include <hip/hip_runtime.h>
#include <hip/hip_bf16.h>

// ---------------------------------------------------------------------------
// RNN classifier for MI455X (gfx1250, wave32, WMMA bf16 path)
//   B=1024, T=128, D=512, QS=100, NA=4, OUTC=10
// 64 workgroups x 16 batch rows; each WG runs all 128 timesteps privately
// (recurrence is independent per batch row -> no inter-WG sync).
// GEMMs: v_wmma_f32_16x16x32_bf16, f32 accumulation; f32 master hidden state
// in LDS; weights prepacked into WMMA B-fragment order (bf16, contiguous k per lane half).
// Each wave computes two N-tiles per GEMM (one A fragment feeds two WMMAs)
// with software-pipelined fragment loads to overlap DS/VMEM with the matrix
// pipe. Classifier is folded into the time loop; softmax at the end.
// ---------------------------------------------------------------------------

typedef __bf16 v16bf __attribute__((ext_vector_type(16)));
typedef __bf16 v4bf  __attribute__((ext_vector_type(4)));
typedef float  v8f   __attribute__((ext_vector_type(8)));

#define B_SZ   1024
#define T_SZ   128
#define D_SZ   512
#define QS     100
#define NA     4
#define OUTC   10
#define NT     7          // 112 = 7 * 16 padded N
#define NPAD   112
#define KT_R   13         // K=400 -> 416 = 13*32   (Wr, Wo)
#define KT_Q   20         // K=612 -> 640 = 20*32   (Wq)
#define KT_X   16         // x part of Wq's K: 512 = 16*32
#define KT_RT  4          // Rt part: 100 -> 128 = 4*32
#define BLK    512        // one 32x16 bf16 fragment block = 512 elements

// ---- dynamic LDS layout (bytes); lAx and lArt contiguous so the Qt GEMM's
// ---- split-K A matrix is a single 20-ktile base pointer.
#define SM_WR   0
#define SM_WO   (SM_WR  + NT*KT_R*BLK*2)     // +93184
#define SM_AX   (SM_WO  + NT*KT_R*BLK*2)     // 186368
#define SM_ART  (SM_AX  + KT_X*BLK*2)        // 202752  (contiguous after lAx)
#define SM_AHA  (SM_ART + KT_RT*BLK*2)       // 206848
#define SM_HA   (SM_AHA + KT_R*BLK*2)        // 220160
#define SM_STG  (SM_HA  + 16*400*4)          // 245760
#define SM_LOG  (SM_STG + 16*NPAD*4)         // 252928
#define SM_TOT  (SM_LOG + 16*OUTC*4)         // 253568 bytes < 320KB

// ---------------------------------------------------------------------------
// Pack row-major W[QS][Ksrc] (f32) into WMMA B-fragment order bf16:
// block (nt, kt) = 32x16 tile B[k,n] = W[n][k]; lane l owns 16 bf16:
// n = nt*16 + (l&15), K = kt*32 + (l>>4)*16 + e.  Zero padded.
// ---------------------------------------------------------------------------
__global__ void pack_b_kernel(const float* __restrict__ W, __bf16* __restrict__ dst,
                              int KT, int Ksrc) {
    int total = NT * KT * BLK;
    for (int p = blockIdx.x * blockDim.x + threadIdx.x; p < total;
         p += gridDim.x * blockDim.x) {
        int blk  = p >> 9;
        int r    = p & 511;
        int lane = r >> 4;
        int e    = r & 15;
        int nt = blk / KT;
        int kt = blk % KT;
        int n  = nt * 16 + (lane & 15);
        int k  = kt * 32 + (lane >> 4) * 16 + e;
        float v = (n < QS && k < Ksrc) ? W[(size_t)n * Ksrc + k] : 0.0f;
        dst[p] = (__bf16)v;
    }
}

// A-fragment position of element (m, k): kt = k/32, kk = k%32,
// e = kk&15, lane = m + 16*(kk>>4).
// For k aligned to 4, the 4 elements k..k+3 are contiguous in one lane.
__device__ __forceinline__ int a_pack_pos(int m, int k) {
    int kt = k >> 5, kk = k & 31;
    int e  = kk & 15;
    int l  = m + ((kk >> 4) << 4);
    return kt * BLK + l * 16 + e;
}

__device__ __forceinline__ v4bf cvt4(float4 v) {
    v4bf h = {(__bf16)v.x, (__bf16)v.y, (__bf16)v.z, (__bf16)v.w};
    return h;
}

// Dual-N-tile K-chain: one A fragment feeds two WMMAs; next-kt fragments are
// fetched before the current WMMAs so loads overlap the matrix pipe.
template<int KT>
__device__ __forceinline__ void gemm_dual(const __bf16* __restrict__ A,
                                          const __bf16* __restrict__ B0,
                                          const __bf16* __restrict__ B1,
                                          int lane, v8f& acc0, v8f& acc1) {
    const int lo = lane * 16;
    v16bf a  = *(const v16bf*)(A  + lo);
    v16bf b0 = *(const v16bf*)(B0 + lo);
    v16bf b1 = *(const v16bf*)(B1 + lo);
#pragma unroll
    for (int kt = 0; kt < KT; ++kt) {
        v16bf an = a, b0n = b0, b1n = b1;
        if (kt + 1 < KT) {
            an  = *(const v16bf*)(A  + (kt + 1) * BLK + lo);
            b0n = *(const v16bf*)(B0 + (kt + 1) * BLK + lo);
            b1n = *(const v16bf*)(B1 + (kt + 1) * BLK + lo);
        }
        acc0 = __builtin_amdgcn_wmma_f32_16x16x32_bf16(
                   false, a, false, b0, (short)0, acc0, false, false);
        acc1 = __builtin_amdgcn_wmma_f32_16x16x32_bf16(
                   false, a, false, b1, (short)0, acc1, false, false);
        a = an; b0 = b0n; b1 = b1n;
    }
}

// C-fragment epilogue: VGPR r holds row m = r + 8*(lane>>4), col n = lane&15.
__device__ __forceinline__ void store_act(float* __restrict__ lStg, const v8f& acc,
                                          int n, int half, float bias) {
#pragma unroll
    for (int r = 0; r < 8; ++r)
        lStg[(r + 8 * half) * NPAD + n] = tanhf(acc[r] + bias);
}

__global__ __launch_bounds__(128)
void rnn_main_kernel(const float* __restrict__ In,
                     const float* __restrict__ br, const float* __restrict__ bq,
                     const float* __restrict__ bo,
                     const float* __restrict__ Wout, const float* __restrict__ bout,
                     const __bf16* __restrict__ gWr, const __bf16* __restrict__ gWo,
                     const __bf16* __restrict__ gWq,
                     float* __restrict__ out) {
    extern __shared__ char smem[];
    __bf16* lWr  = (__bf16*)(smem + SM_WR);
    __bf16* lWo  = (__bf16*)(smem + SM_WO);
    __bf16* lAx  = (__bf16*)(smem + SM_AX);   // also Qt A base (20 ktiles)
    __bf16* lArt = (__bf16*)(smem + SM_ART);
    __bf16* lAha = (__bf16*)(smem + SM_AHA);
    float*  lHa  = (float* )(smem + SM_HA);
    float*  lStg = (float* )(smem + SM_STG);
    float*  lLog = (float* )(smem + SM_LOG);

    const int tid  = threadIdx.x;
    const int lane = tid & 31;          // wave32
    const int wave = tid >> 5;          // 4 waves per WG
    const int m0   = blockIdx.x * 16;   // this WG's batch rows

    // wave's two N-tiles (wave 3 duplicates nt0; its acc1 store is skipped)
    const int nt0 = wave;
    const int nt1 = (wave + 4 < NT) ? wave + 4 : wave;
    const int ncol  = lane & 15;
    const int half  = lane >> 4;
    const int n0 = nt0 * 16 + ncol;
    const int n1 = nt1 * 16 + ncol;

    // time-invariant per-lane biases (hoisted out of the hot loop)
    const float br0 = (n0 < QS) ? br[n0] : 0.0f;
    const float br1 = (n1 < QS) ? br[n1] : 0.0f;
    const float bq0 = (n0 < QS) ? bq[n0] : 0.0f;
    const float bq1 = (n1 < QS) ? bq[n1] : 0.0f;
    const float bo0 = (n0 < QS) ? bo[n0] : 0.0f;
    const float bo1 = (n1 < QS) ? bo[n1] : 0.0f;

    // ---- preload resident packed weights (Wr, Wo) into LDS, b128 copies ---
    {
        const uint4* s1 = (const uint4*)gWr;
        const uint4* s2 = (const uint4*)gWo;
        uint4* d1 = (uint4*)lWr;
        uint4* d2 = (uint4*)lWo;
        const int n16 = NT * KT_R * BLK * 2 / 16;
        for (int i = tid; i < n16; i += blockDim.x) { d1[i] = s1[i]; d2[i] = s2[i]; }
    }
    // ---- zero state (ha = 0, incl. K-pad regions of A buffers, logits) ----
    for (int i = tid; i < 16 * 400;   i += blockDim.x) lHa[i]  = 0.0f;
    for (int i = tid; i < KT_R * BLK; i += blockDim.x) lAha[i] = (__bf16)0.0f;
    for (int i = tid; i < KT_RT* BLK; i += blockDim.x) lArt[i] = (__bf16)0.0f;
    for (int i = tid; i < 16 * OUTC;  i += blockDim.x) lLog[i] = 0.0f;
    __syncthreads();

    for (int t = 0; t < T_SZ; ++t) {
        // -- Phase A: x_t (16 x 512) f32 -> bf16 A-fragments, float4 wide ---
        for (int idx = tid; idx < 16 * 128; idx += blockDim.x) {
            int m  = idx >> 7;
            int k4 = (idx & 127) << 2;
            const float4* src =
                (const float4*)&In[((size_t)(m0 + m) * T_SZ + t) * D_SZ + k4];
            *(v4bf*)&lAx[a_pack_pos(m, k4)] = cvt4(*src);
            if ((k4 & 63) == 0 && t + 1 < T_SZ)
                __builtin_prefetch((const float*)src + D_SZ, 0, 1);
        }
        __syncthreads();

        // -- Phase B: Rt = tanh(ha @ Wr^T + br)  (K=416, LDS-resident B) ----
        {
            v8f acc0 = {}, acc1 = {};
            gemm_dual<KT_R>(lAha, lWr + nt0 * KT_R * BLK, lWr + nt1 * KT_R * BLK,
                            lane, acc0, acc1);
            store_act(lStg, acc0, n0, half, br0);
            if (nt1 != nt0) store_act(lStg, acc1, n1, half, br1);
        }
        __syncthreads();

        // -- Phase C: repack Rt into A-fragments (K'=128, zero pad) ---------
        for (int idx = tid; idx < 16 * 32; idx += blockDim.x) {
            int m  = idx >> 5;
            int k4 = (idx & 31) << 2;
            v4bf h = {(__bf16)0.0f, (__bf16)0.0f, (__bf16)0.0f, (__bf16)0.0f};
            if (k4 < QS) h = cvt4(*(const float4*)&lStg[m * NPAD + k4]);
            *(v4bf*)&lArt[a_pack_pos(m, k4)] = h;
        }
        __syncthreads();

        // -- Phase D: Qt = tanh([x_t,Rt] @ Wq^T + bq)  (K=640) --------------
        //    A: lAx..lArt contiguous (20 ktiles); B streamed from L2 Wq.
        {
            v8f acc0 = {}, acc1 = {};
            gemm_dual<KT_Q>(lAx, gWq + (size_t)nt0 * KT_Q * BLK,
                                 gWq + (size_t)nt1 * KT_Q * BLK,
                            lane, acc0, acc1);
            store_act(lStg, acc0, n0, half, bq0);
            if (nt1 != nt0) store_act(lStg, acc1, n1, half, bq1);
        }
        __syncthreads();

        // -- Phase E: leaky integration in f32, refresh bf16 A-fragments ----
        //    groups of 4 never straddle an alpha segment (100 % 4 == 0)
        for (int idx = tid; idx < 16 * 100; idx += blockDim.x) {
            int m  = idx / 100;
            int h4 = (idx % 100) << 2;          // 0..396
            int na = h4 / QS;
            int q  = h4 % QS;
            float alpha = (na == 0) ? 0.0f : (na == 1) ? 0.25f
                        : (na == 2) ? 0.5f : 0.95f;
            float beta = 1.0f - alpha;
            float4 ha = *(float4*)&lHa[m * 400 + h4];
            float4 qt = *(const float4*)&lStg[m * NPAD + q];
            ha.x = alpha * ha.x + beta * qt.x;
            ha.y = alpha * ha.y + beta * qt.y;
            ha.z = alpha * ha.z + beta * qt.z;
            ha.w = alpha * ha.w + beta * qt.w;
            *(float4*)&lHa[m * 400 + h4] = ha;
            *(v4bf*)&lAha[a_pack_pos(m, h4)] = cvt4(ha);
        }
        __syncthreads();

        // -- Phase F: out_t = tanh(ha' @ Wo^T + bo) -------------------------
        {
            v8f acc0 = {}, acc1 = {};
            gemm_dual<KT_R>(lAha, lWo + nt0 * KT_R * BLK, lWo + nt1 * KT_R * BLK,
                            lane, acc0, acc1);
            store_act(lStg, acc0, n0, half, bo0);
            if (nt1 != nt0) store_act(lStg, acc1, n1, half, bo1);
        }
        __syncthreads();

        // -- Phase G: logits += out_t @ Wout[:, t*100 : t*100+100].T --------
        for (int p = tid; p < 16 * OUTC; p += blockDim.x) {
            int m = p / OUTC, c = p % OUTC;
            const float4* w4 = (const float4*)(Wout + (size_t)c * (QS * T_SZ) + t * QS);
            const float4* s4 = (const float4*)(lStg + m * NPAD);
            float s = 0.0f;
            for (int i = 0; i < QS / 4; ++i) {
                float4 a = s4[i], b = w4[i];
                s += a.x * b.x + a.y * b.y + a.z * b.z + a.w * b.w;
            }
            lLog[p] += s;       // single writer per (m,c)
        }
        __syncthreads();
    }

    // ---- softmax over 10 classes, one thread per batch row ----------------
    if (tid < 16) {
        int m = tid;
        float lv[OUTC];
        float mx = -1e30f;
        for (int c = 0; c < OUTC; ++c) {
            lv[c] = lLog[m * OUTC + c] + bout[c];
            mx = fmaxf(mx, lv[c]);
        }
        float s = 0.0f;
        for (int c = 0; c < OUTC; ++c) { lv[c] = expf(lv[c] - mx); s += lv[c]; }
        float inv = 1.0f / s;
        for (int c = 0; c < OUTC; ++c)
            out[(size_t)(m0 + m) * OUTC + c] = lv[c] * inv;
    }
}

extern "C" void kernel_launch(void* const* d_in, const int* in_sizes, int n_in,
                              void* d_out, int out_size, void* d_ws, size_t ws_size,
                              hipStream_t stream) {
    (void)in_sizes; (void)n_in; (void)out_size; (void)ws_size;

    const float* In   = (const float*)d_in[0];   // (1024,128,512)
    const float* Wr   = (const float*)d_in[1];   // (100,400)
    const float* br   = (const float*)d_in[2];
    const float* Wq   = (const float*)d_in[3];   // (100,612)
    const float* bq   = (const float*)d_in[4];
    const float* Wo   = (const float*)d_in[5];   // (100,400)
    const float* bo   = (const float*)d_in[6];
    const float* Wout = (const float*)d_in[7];   // (10,12800)
    const float* bout = (const float*)d_in[8];
    float* out = (float*)d_out;

    // workspace: packed bf16 weights (total ~322 KB, stays L2-resident)
    __bf16* pWr = (__bf16*)d_ws;
    __bf16* pWo = pWr + (size_t)NT * KT_R * BLK;
    __bf16* pWq = pWo + (size_t)NT * KT_R * BLK;

    pack_b_kernel<<<64, 256, 0, stream>>>(Wr, pWr, KT_R, 400);
    pack_b_kernel<<<64, 256, 0, stream>>>(Wo, pWo, KT_R, 400);
    pack_b_kernel<<<64, 256, 0, stream>>>(Wq, pWq, KT_Q, 612);

    // 247 KB dynamic LDS per workgroup (WGP has 320 KB)
    (void)hipFuncSetAttribute((const void*)rnn_main_kernel,
                              hipFuncAttributeMaxDynamicSharedMemorySize, SM_TOT);
    rnn_main_kernel<<<B_SZ / 16, 128, SM_TOT, stream>>>(
        In, br, bq, bo, Wout, bout, pWr, pWo, pWq, out);
}
